// ContrastiveLoss_20959440404716
// MI455X (gfx1250) — matrix operand order repaired and checked by reading, not compiled, not run on hardware
//
#include <hip/hip_runtime.h>
#include <hip/hip_bf16.h>

typedef _Float16 h16 __attribute__((ext_vector_type(16)));
typedef _Float16 h8  __attribute__((ext_vector_type(8)));
typedef float    f8  __attribute__((ext_vector_type(8)));
typedef unsigned int u32x4 __attribute__((ext_vector_type(4)));
typedef int          i32x4 __attribute__((ext_vector_type(4)));
typedef int          i32x8 __attribute__((ext_vector_type(8)));

#define N_ROWS 8192
#define DIM    256
#define NTILES 512            // 8192/16 column tiles
#define NSEG   8
#define TPS    (NTILES/NSEG)  // 64 tiles per segment
#define WAVES  8              // waves (row-blocks) per workgroup
#define INV_T  10.0f

__device__ __forceinline__ h16 cat16(h8 a, h8 b) {
    return __builtin_shufflevector(a, b, 0,1,2,3,4,5,6,7,8,9,10,11,12,13,14,15);
}

// TDM: DMA one 16-row x 512-byte tile (16 logit-columns x 256 f16) into LDS.
// D# group0: count=1, lds_addr, 57b global_addr, type=2.
// D# group1: data_size=8B, tensor_dim0=64 units, tensor_dim1=8192 rows,
//            tile_dim0=64, tile_dim1=16, dim0_stride=64 (contiguous rows).
__device__ __forceinline__ void tdm_load_tile(const _Float16* src_row0, unsigned lds_off) {
    unsigned long long ga = (unsigned long long)(uintptr_t)src_row0;
    u32x4 g0 = { 1u, lds_off, (unsigned)ga,
                 (unsigned)((ga >> 32) & 0x01ffffffull) | 0x80000000u };
    i32x8 g1 = { (int)(3u << 16),      // data_size = 8 bytes
                 (int)(64u << 16),     // tensor_dim0 (low16 @ bits 63:48)
                 (int)(8192u << 16),   // tensor_dim0 hi=0 | tensor_dim1 low16
                 (int)(64u << 16),     // tensor_dim1 hi=0 | tile_dim0=64
                 16,                   // tile_dim1=16 | tile_dim2=0
                 64,                   // tensor_dim0_stride low32
                 0, 0 };               // stride hi / dim1_stride = 0
    i32x4 gz = { 0, 0, 0, 0 };
#if defined(__clang_major__) && __clang_major__ >= 23
    i32x8 gz8 = { 0,0,0,0,0,0,0,0 };
    __builtin_amdgcn_tensor_load_to_lds(g0, g1, gz, gz, gz8, 0);
#else
    __builtin_amdgcn_tensor_load_to_lds(g0, g1, gz, gz, 0);
#endif
}

// ---------------- Phase A: row L2-normalize + fp16 hi/lo split ----------------
__global__ void ContrastiveLoss_normalize(const float* __restrict__ z,
                                          _Float16* __restrict__ zhi,
                                          _Float16* __restrict__ zlo,
                                          float* __restrict__ s_arr,
                                          float* __restrict__ out) {
    const int row = blockIdx.x;
    const int t   = threadIdx.x;            // 256 threads = one row
    float x  = z[row * DIM + t];
    float sq = x * x;
    #pragma unroll
    for (int off = 16; off >= 1; off >>= 1) sq += __shfl_xor(sq, off, 32);
    __shared__ float part[8];
    if ((t & 31) == 0) part[t >> 5] = sq;
    __syncthreads();
    float tot = 0.f;
    #pragma unroll
    for (int i = 0; i < 8; ++i) tot += part[i];
    const float r = 1.0f / fmaxf(sqrtf(tot), 1e-8f);
    const float v = x * r;
    const _Float16 h = (_Float16)v;
    zhi[row * DIM + t] = h;
    zlo[row * DIM + t] = (_Float16)(v - (float)h);
    if (t == 0) s_arr[row] = 0.f;                 // zero exp-sum accumulators
    if (row == 0 && t == 0) *out = 0.f;           // zero the scalar output
}

// ---------------- Phase B: TDM-staged, fused WMMA GEMM + exp-sum --------------
__global__ void __launch_bounds__(WAVES * 32)
ContrastiveLoss_tiles(const _Float16* __restrict__ zhi,
                      const _Float16* __restrict__ zlo,
                      float* __restrict__ s_arr,
                      float* __restrict__ p_arr) {
    const int tid  = threadIdx.x;
    const int wave = tid >> 5;
    const int lane = tid & 31;
    const int rowblock = blockIdx.x * WAVES + wave;   // 0..511
    const int seg      = blockIdx.y;                  // 0..7
    const int rowbase  = rowblock * 16;
    const int j        = lane & 15;
    const bool lowhalf = lane < 16;
    const int selftile = rowblock;
    const int postile  = (rowblock + 256) & (NTILES - 1);

    // Double-buffered B tiles: [buf][hi/lo][16 cols x 256 f16] = 32 KB
    __shared__ _Float16 sbuf[2][2][16 * DIM];

    // --- preload A fragments (hi + lo) for this wave's 16 rows, K = 0..255 ---
    // ISA A layout (f16 16x32): lanes0-15 hold K {0..7,16..23}, lanes16-31 {8..15,24..31}
    const _Float16* arh = zhi + (rowbase + j) * DIM;
    const _Float16* arl = zlo + (rowbase + j) * DIM;
    h16 a_hi[8], a_lo[8];
    #pragma unroll
    for (int k = 0; k < 8; ++k) {
        const int kb = k * 32;
        const int o1 = kb + (lowhalf ? 0 : 8);
        const int o2 = kb + (lowhalf ? 16 : 24);
        a_hi[k] = cat16(*(const h8*)(arh + o1), *(const h8*)(arh + o2));
        a_lo[k] = cat16(*(const h8*)(arl + o1), *(const h8*)(arl + o2));
    }

    float s[8], p[8];
    #pragma unroll
    for (int r = 0; r < 8; ++r) { s[r] = 0.f; p[r] = 0.f; }

    // B layout (f16 32x16): lane = column (j), lanes0-15 K {0..7,16..23}, lanes16-31 K {8..15,24..31}
    const int koff = lowhalf ? 0 : 8;
    const int ct0  = seg * TPS;

    // Prologue: wave0 DMAs the first tile into buffer 0.
    if (wave == 0) {
        tdm_load_tile(zhi + (size_t)(ct0 * 16) * DIM, (unsigned)(uintptr_t)&sbuf[0][0][0]);
        tdm_load_tile(zlo + (size_t)(ct0 * 16) * DIM, (unsigned)(uintptr_t)&sbuf[0][1][0]);
    }

    int cur = 0;
    for (int it = 0; it < TPS; ++it) {
        if (wave == 0) __builtin_amdgcn_s_wait_tensorcnt(0);
        __syncthreads();                       // buf[cur] ready; buf[cur^1] free
        if (wave == 0 && it + 1 < TPS) {
            const int nc = (ct0 + it + 1) * 16;
            tdm_load_tile(zhi + (size_t)nc * DIM, (unsigned)(uintptr_t)&sbuf[cur ^ 1][0][0]);
            tdm_load_tile(zlo + (size_t)nc * DIM, (unsigned)(uintptr_t)&sbuf[cur ^ 1][1][0]);
        }

        const _Float16* bh = &sbuf[cur][0][j * DIM + koff];
        const _Float16* bl = &sbuf[cur][1][j * DIM + koff];

        // Two independent accumulator chains: c0 = hi*hi, c1 = hi*lo + lo*hi
        f8 c0 = {}, c1 = {};
        #pragma unroll
        for (int k = 0; k < 8; ++k) {
            h16 bhv = cat16(*(const h8*)(bh + k * 32), *(const h8*)(bh + k * 32 + 16));
            h16 blv = cat16(*(const h8*)(bl + k * 32), *(const h8*)(bl + k * 32 + 16));
            c0 = __builtin_amdgcn_wmma_f32_16x16x32_f16(false, a_hi[k], false, bhv, (short)0, c0, false, false);
            c1 = __builtin_amdgcn_wmma_f32_16x16x32_f16(false, a_hi[k], false, blv, (short)0, c1, false, false);
            c1 = __builtin_amdgcn_wmma_f32_16x16x32_f16(false, a_lo[k], false, bhv, (short)0, c1, false, false);
        }

        const int ct     = ct0 + it;
        const bool isself = (ct == selftile);
        const bool ispos  = (ct == postile);
        #pragma unroll
        for (int r = 0; r < 8; ++r) {
            // C layout: lane<16 -> (M=r, N=j); lane>=16 -> (M=8+r, N=j)
            const bool diag = lowhalf ? (j == r) : (j == r + 8);
            const float v = (c0[r] + c1[r]) * INV_T;   // logit, bounded by ~10
            float e = __expf(v - 10.0f);               // fixed-max logsumexp
            if (isself && diag) e = 0.f;               // drop self-similarity
            s[r] += e;
            if (ispos && diag) p[r] = v;               // capture positive logit
        }
        cur ^= 1;
    }

    // reduce exp-sums / positives across the 16 lanes of each half-wave
    #pragma unroll
    for (int r = 0; r < 8; ++r) {
        #pragma unroll
        for (int off = 8; off >= 1; off >>= 1) {
            s[r] += __shfl_xor(s[r], off, 32);
            p[r] += __shfl_xor(p[r], off, 32);
        }
    }

    if (j == 0) {
        const int rb = rowbase + (lowhalf ? 0 : 8);
        const bool own_pos = (postile / TPS) == seg;
        #pragma unroll
        for (int r = 0; r < 8; ++r) {
            atomicAdd(&s_arr[rb + r], s[r]);
            if (own_pos) p_arr[rb + r] = p[r];
        }
    }
}

// ---------------- Phase C: loss = -mean(p - 10 - log(s)) ----------------------
__global__ void ContrastiveLoss_finalize(const float* __restrict__ s_arr,
                                         const float* __restrict__ p_arr,
                                         float* __restrict__ out) {
    const int row = blockIdx.x * 256 + threadIdx.x;
    const float lp = p_arr[row] - 10.0f - __logf(s_arr[row]);
    float contrib = -lp * (1.0f / (float)N_ROWS);
    #pragma unroll
    for (int off = 16; off >= 1; off >>= 1) contrib += __shfl_xor(contrib, off, 32);
    if ((threadIdx.x & 31) == 0) atomicAdd(out, contrib);
}

extern "C" void kernel_launch(void* const* d_in, const int* in_sizes, int n_in,
                              void* d_out, int out_size, void* d_ws, size_t ws_size,
                              hipStream_t stream) {
    const float* z = (const float*)d_in[0];
    float* out = (float*)d_out;
    char* ws = (char*)d_ws;

    _Float16* zhi  = (_Float16*)ws;                                   // 4 MB
    _Float16* zlo  = (_Float16*)(ws + (size_t)N_ROWS * DIM * 2);      // 4 MB
    float*    s_arr = (float*)(ws + (size_t)N_ROWS * DIM * 4);        // 32 KB
    float*    p_arr = s_arr + N_ROWS;                                 // 32 KB

    ContrastiveLoss_normalize<<<N_ROWS, 256, 0, stream>>>(z, zhi, zlo, s_arr, out);
    dim3 grid(NTILES / WAVES, NSEG);
    ContrastiveLoss_tiles<<<grid, WAVES * 32, 0, stream>>>(zhi, zlo, s_arr, p_arr);
    ContrastiveLoss_finalize<<<N_ROWS / 256, 256, 0, stream>>>(s_arr, p_arr, out);
}
